// MultiHeadAttention_75496935129470
// MI455X (gfx1250) — matrix operand order repaired and checked by reading, not compiled, not run on hardware
//
#include <hip/hip_runtime.h>
#include <hip/hip_bf16.h>

// ---------------------------------------------------------------------------
// MHA forward for MI455X (gfx1250, wave32, WMMA f32_16x16x32_f16)
// B=2, S=2048, E=1024, H=16, D=64
// ---------------------------------------------------------------------------

typedef __attribute__((ext_vector_type(16))) _Float16 v16h;
typedef __attribute__((ext_vector_type(8)))  float    v8f;

constexpr int Bc = 2, Sc = 2048, Ec = 1024, Hc = 16, Dc = 64;
constexpr int NTOK = Bc * Sc;  // 4096 token rows

union AFrag { v16h v; uint4 u[2]; _Float16 h[16]; };

// A-matrix fragment (16x32 f16): lane holds row M=lane&15.
// Halves 0..7  = K (k0 + koff + 0..7),  halves 8..15 = K (k0 + koff + 16..23)
// with koff = (lane>=16) ? 8 : 0.   (ISA 7.12.2, 16-bit A 16x32 layout)
__device__ __forceinline__ v16h load_a_frag(const _Float16* rowptr, int k0) {
  int lane = threadIdx.x & 31;
  int koff = (lane >= 16) ? 8 : 0;
  AFrag f;
  f.u[0] = *(const uint4*)(rowptr + k0 + koff);
  f.u[1] = *(const uint4*)(rowptr + k0 + koff + 16);
  return f.v;
}

// B-matrix fragment (32x16 f16): lane holds column N=lane&15.
// Halves 0..7 = K (kbase + 8*(lane>=16) + 0..7), halves 8..15 = K (kbase + 8*(lane>=16) + 16..23).
__device__ __forceinline__ v16h load_b_frag(const _Float16* rowptr, int kbase) {
  int lane = threadIdx.x & 31;
  int off = kbase + ((lane >= 16) ? 8 : 0);
  AFrag f;
  f.u[0] = *(const uint4*)(rowptr + off);
  f.u[1] = *(const uint4*)(rowptr + off + 16);
  return f.v;
}

// RA/RB as template params so the builtin's reuse-hint immediates are ICEs.
// reuse_a is set on a WMMA whose *next* instruction shares the same A operand.
template <bool RA = false, bool RB = false>
__device__ __forceinline__ v8f wmma_f16(v16h a, v16h b, v8f c) {
  // (neg_a, A, neg_b, B, c_mod, C, reuse_a, reuse_b)
  return __builtin_amdgcn_wmma_f32_16x16x32_f16(false, a, false, b,
                                                (short)0, c, RA, RB);
}

// ---------------------------------------------------------------------------
// f32 -> f16 conversion (4 elements/thread, vectorized loads/stores)
// ---------------------------------------------------------------------------
__global__ void __launch_bounds__(256)
cvt_f32_f16_kernel(const float* __restrict__ src, _Float16* __restrict__ dst,
                   int n) {
  int i = (blockIdx.x * blockDim.x + threadIdx.x) * 4;
  if (i + 4 > n) return;
  float4 f = *(const float4*)(src + i);
  union { uint2 u; _Float16 h[4]; } o;
  o.h[0] = (_Float16)f.x; o.h[1] = (_Float16)f.y;
  o.h[2] = (_Float16)f.z; o.h[3] = (_Float16)f.w;
  *(uint2*)(dst + i) = o.u;
}

// ---------------------------------------------------------------------------
// Register-blocked GEMM: out = X[M x E] @ W^T + bias, 64x64 tile per wave.
// Per k-step (k += 32): 4 A-frags + 4 B-frags (16x global_load_b128) feed
// 16 WMMAs -> 1:1 wmma:load issue instead of 1:4. ~210 VGPRs/lane (~4
// waves/SIMD), K loop = 32 iters -> 512 WMMA/wave.
// W kept in [Ncols x E] row-major: B-fragment of W^T is a contiguous row read.
// mode 0: f16 out, head layout  [B,H,S,D]   (Q/K projections)
// mode 1: f16 out, head layout  [B,H,D,S]   (V projection, transposed)
// mode 2: f32 out, row-major    [M,E]       (final output projection)
// ---------------------------------------------------------------------------
__global__ void __launch_bounds__(256)
gemm_wmma_kernel(const _Float16* __restrict__ X, const _Float16* __restrict__ W,
                 const float* __restrict__ bias, void* __restrict__ out,
                 int Mrows, int mode) {
  int wave = threadIdx.x >> 5;
  int lane = threadIdx.x & 31;
  int tile = blockIdx.x * (blockDim.x >> 5) + wave;
  const int tilesN = Ec / 64;  // 16
  int tm = tile / tilesN, tn = tile % tilesN;
  int m0 = tm * 64, n0 = tn * 64;
  if (m0 >= Mrows) return;

  // Row base pointers; the i*16 / j*16 row steps are constant byte offsets
  // (32 KB) that fold into the 24-bit instruction offset.
  const _Float16* arow = X + (long)(m0 + (lane & 15)) * Ec;
  const _Float16* brow = W + (long)(n0 + (lane & 15)) * Ec;

  v8f acc[4][4];  // [j: N subtile][i: M subtile]
#pragma unroll
  for (int j = 0; j < 4; ++j)
#pragma unroll
    for (int i = 0; i < 4; ++i) acc[j][i] = (v8f){};

  for (int k0 = 0; k0 < Ec; k0 += 32) {
    v16h af[4], bf[4];
#pragma unroll
    for (int i = 0; i < 4; ++i) af[i] = load_a_frag(arow + i * 16 * Ec, k0);
#pragma unroll
    for (int j = 0; j < 4; ++j) bf[j] = load_b_frag(brow + j * 16 * Ec, k0);
#pragma unroll
    for (int i = 0; i < 4; ++i) {
      // 4 consecutive WMMAs share af[i]: set reuse-A on the first three.
      acc[0][i] = wmma_f16<true>(af[i], bf[0], acc[0][i]);
      acc[1][i] = wmma_f16<true>(af[i], bf[1], acc[1][i]);
      acc[2][i] = wmma_f16<true>(af[i], bf[2], acc[2][i]);
      acc[3][i] = wmma_f16<false>(af[i], bf[3], acc[3][i]);
    }
  }

  int rowoff = (lane >= 16) ? 8 : 0;
#pragma unroll
  for (int j = 0; j < 4; ++j) {
    int ncol = n0 + j * 16 + (lane & 15);
    float bv = bias[ncol];
    int h = ncol / Dc, d = ncol % Dc;  // h == tn*64/Dc region, d in [0,64)
#pragma unroll
    for (int i = 0; i < 4; ++i) {
#pragma unroll
      for (int r = 0; r < 8; ++r) {
        int n = m0 + i * 16 + rowoff + r;  // token row
        float val = acc[j][i][r] + bv;
        if (mode == 2) {
          ((float*)out)[(long)n * Ec + ncol] = val;
        } else {
          int b = n / Sc, s = n % Sc;  // S multiple of 64: tiles never straddle
          long idx = (mode == 0)
                         ? ((long)((b * Hc + h) * Sc + s)) * Dc + d
                         : ((long)((b * Hc + h) * Dc + d)) * Sc + s;
          ((_Float16*)out)[idx] = (_Float16)val;
        }
      }
    }
  }
}

// ---------------------------------------------------------------------------
// Flash attention: one wave per 16-query tile of one (b,h).
// Iterates keys 32 at a time: 4 WMMAs for scores (2 subtiles x K=64 contraction
// in 2 steps), online softmax (shfl_xor row reductions across the 16-lane
// half-groups), P transposed C-layout -> A-layout via LDS (+ s_wait_dscnt),
// then 4 WMMAs accumulating ctx over the 4 d-tiles of D=64.
// ---------------------------------------------------------------------------
#define ATT_WAVES 8

__global__ void __launch_bounds__(256)
flash_attn_kernel(const _Float16* __restrict__ qh,  // [B*H,S,D]
                  const _Float16* __restrict__ kh,  // [B*H,S,D]
                  const _Float16* __restrict__ vt,  // [B*H,D,S]
                  const int* __restrict__ mask,     // [B,S,S]
                  _Float16* __restrict__ ctx) {     // [B,S,E]
  __shared__ __align__(16) _Float16 Pbuf[ATT_WAVES][16 * 32];
  int wave = threadIdx.x >> 5;
  int lane = threadIdx.x & 31;
  int qt = blockIdx.x * ATT_WAVES + wave;  // global q-tile (4096 total)
  const int tilesPerBH = Sc / 16;          // 128
  int bh = qt / tilesPerBH;
  int q0 = (qt % tilesPerBH) * 16;
  int b = bh / Hc, h = bh % Hc;
  int rowoff = (lane >= 16) ? 8 : 0;

  // Q fragments for the whole tile (D=64 -> two 16x32 A-fragments), kept live.
  const _Float16* qrow = qh + ((long)bh * Sc + q0 + (lane & 15)) * Dc;
  v16h aq0 = load_a_frag(qrow, 0);
  v16h aq1 = load_a_frag(qrow, 32);

  v8f c0 = {}, c1 = {}, c2 = {}, c3 = {};  // ctx accumulators, 16 x 64 f32
  float m[8], l[8];
#pragma unroll
  for (int r = 0; r < 8; ++r) { m[r] = -3.0e38f; l[r] = 0.0f; }

  const int* maskbase = mask + (long)b * Sc * Sc;
  _Float16* P = Pbuf[wave];
  const float scale = 0.125f;  // 1/sqrt(D), D=64
  v8f zero = {};

  for (int k0 = 0; k0 < Sc; k0 += 32) {
    if (k0 + 32 < Sc)  // global_prefetch_b8 for the next K tile
      __builtin_prefetch(kh + ((long)bh * Sc + k0 + 32 + (lane & 15)) * Dc, 0, 1);

    // ---- scores: S = Q @ K^T for two 16-key subtiles --------------------
    const _Float16* krowA = kh + ((long)bh * Sc + k0 + (lane & 15)) * Dc;
    const _Float16* krowB = krowA + 16 * Dc;
    // Ordered so consecutive WMMAs share the A operand (reuse-A hints).
    v8f t0A = wmma_f16<true>(aq0, load_b_frag(krowA, 0), zero);
    v8f t0B = wmma_f16<false>(aq0, load_b_frag(krowB, 0), zero);
    v8f sA = wmma_f16<true>(aq1, load_b_frag(krowA, 32), t0A);
    v8f sB = wmma_f16<false>(aq1, load_b_frag(krowB, 32), t0B);

    // ---- scale + mask ----------------------------------------------------
#pragma unroll
    for (int r = 0; r < 8; ++r) {
      int q = q0 + rowoff + r;
      long mi = (long)q * Sc + k0 + (lane & 15);
      sA[r] = (maskbase[mi] == 0)      ? -1.0e10f : sA[r] * scale;
      sB[r] = (maskbase[mi + 16] == 0) ? -1.0e10f : sB[r] * scale;
    }

    // ---- online softmax per row (rows live in VGPR components) -----------
#pragma unroll
    for (int r = 0; r < 8; ++r) {
      float t = fmaxf(sA[r], sB[r]);
#pragma unroll
      for (int xm = 1; xm <= 8; xm <<= 1) t = fmaxf(t, __shfl_xor(t, xm, 32));
      float mnew = fmaxf(m[r], t);
      float pA = __expf(sA[r] - mnew);
      float pB = __expf(sB[r] - mnew);
      float alpha = __expf(m[r] - mnew);
      float rs = pA + pB;
#pragma unroll
      for (int xm = 1; xm <= 8; xm <<= 1) rs += __shfl_xor(rs, xm, 32);
      l[r] = l[r] * alpha + rs;
      m[r] = mnew;
      c0[r] *= alpha; c1[r] *= alpha; c2[r] *= alpha; c3[r] *= alpha;
      // stage P tile (row-major 16x32) in this wave's private LDS slice
      P[(rowoff + r) * 32 + (lane & 15)]      = (_Float16)pA;
      P[(rowoff + r) * 32 + 16 + (lane & 15)] = (_Float16)pB;
    }
    // cross-lane LDS RAW inside the wave: drain DS counter explicitly
    asm volatile("s_wait_dscnt 0" ::: "memory");

    // ---- P (A-layout via ds_load_b128) and ctx += P @ V ------------------
    v16h ap = load_a_frag(P + (lane & 15) * 32, 0);
    const _Float16* vbase = vt + ((long)bh * Dc + (lane & 15)) * Sc + k0;
    c0 = wmma_f16<true>(ap, load_b_frag(vbase + (long)0 * 16 * Sc, 0), c0);
    c1 = wmma_f16<true>(ap, load_b_frag(vbase + (long)1 * 16 * Sc, 0), c1);
    c2 = wmma_f16<true>(ap, load_b_frag(vbase + (long)2 * 16 * Sc, 0), c2);
    c3 = wmma_f16<false>(ap, load_b_frag(vbase + (long)3 * 16 * Sc, 0), c3);
  }

  // ---- normalize and store ctx in [B,S,E] f16 for the output GEMM ---------
#pragma unroll
  for (int r = 0; r < 8; ++r) {
    float inv = 1.0f / l[r];
    int q = q0 + rowoff + r;
    long o = ((long)b * Sc + q) * Ec + (long)h * Dc + (lane & 15);
    ctx[o + 0]  = (_Float16)(c0[r] * inv);
    ctx[o + 16] = (_Float16)(c1[r] * inv);
    ctx[o + 32] = (_Float16)(c2[r] * inv);
    ctx[o + 48] = (_Float16)(c3[r] * inv);
  }
}

// ---------------------------------------------------------------------------
extern "C" void kernel_launch(void* const* d_in, const int* in_sizes, int n_in,
                              void* d_out, int out_size, void* d_ws, size_t ws_size,
                              hipStream_t stream) {
  const float* q    = (const float*)d_in[0];
  const float* k    = (const float*)d_in[1];
  const float* v    = (const float*)d_in[2];
  const int*   mask = (const int*)  d_in[3];
  const float* Wq   = (const float*)d_in[4];
  const float* bq   = (const float*)d_in[5];
  const float* Wk   = (const float*)d_in[6];
  const float* bk   = (const float*)d_in[7];
  const float* Wv   = (const float*)d_in[8];
  const float* bv   = (const float*)d_in[9];
  const float* Wo   = (const float*)d_in[10];
  const float* bo   = (const float*)d_in[11];
  float* out = (float*)d_out;

  const size_t MB = 1u << 20;
  char* ws = (char*)d_ws;
  _Float16* Xq   = (_Float16*)(ws + 0 * MB);   // 4M halves each
  _Float16* Xk   = (_Float16*)(ws + 8 * MB);
  _Float16* Xv   = (_Float16*)(ws + 16 * MB);
  _Float16* Wq16 = (_Float16*)(ws + 24 * MB);  // 1M halves each
  _Float16* Wk16 = (_Float16*)(ws + 26 * MB);
  _Float16* Wv16 = (_Float16*)(ws + 28 * MB);
  _Float16* Wo16 = (_Float16*)(ws + 30 * MB);
  _Float16* qhB  = (_Float16*)(ws + 32 * MB);  // [B,H,S,D]
  _Float16* khB  = (_Float16*)(ws + 40 * MB);  // [B,H,S,D]
  _Float16* vtB  = (_Float16*)(ws + 48 * MB);  // [B,H,D,S]
  _Float16* ctxB = (_Float16*)(ws + 56 * MB);  // [B,S,E]

  const int nX = NTOK * Ec;  // 4,194,304
  const int nW = Ec * Ec;    // 1,048,576
  dim3 blk(256);
  cvt_f32_f16_kernel<<<nX / 1024, blk, 0, stream>>>(q,  Xq,   nX);
  cvt_f32_f16_kernel<<<nX / 1024, blk, 0, stream>>>(k,  Xk,   nX);
  cvt_f32_f16_kernel<<<nX / 1024, blk, 0, stream>>>(v,  Xv,   nX);
  cvt_f32_f16_kernel<<<nW / 1024, blk, 0, stream>>>(Wq, Wq16, nW);
  cvt_f32_f16_kernel<<<nW / 1024, blk, 0, stream>>>(Wk, Wk16, nW);
  cvt_f32_f16_kernel<<<nW / 1024, blk, 0, stream>>>(Wv, Wv16, nW);
  cvt_f32_f16_kernel<<<nW / 1024, blk, 0, stream>>>(Wo, Wo16, nW);

  // projections: (4096/64) * (1024/64) = 1024 tiles, 8 waves/block -> 128 blks
  const int gemmBlocks = (NTOK / 64) * (Ec / 64) / 8;
  gemm_wmma_kernel<<<gemmBlocks, blk, 0, stream>>>(Xq, Wq16, bq, qhB, NTOK, 0);
  gemm_wmma_kernel<<<gemmBlocks, blk, 0, stream>>>(Xk, Wk16, bk, khB, NTOK, 0);
  gemm_wmma_kernel<<<gemmBlocks, blk, 0, stream>>>(Xv, Wv16, bv, vtB, NTOK, 1);

  // attention: B*H*(S/16) = 4096 q-tiles, 8 waves/block -> 512 blocks
  flash_attn_kernel<<<(Bc * Hc * (Sc / 16)) / ATT_WAVES, blk, 0, stream>>>(
      qhB, khB, vtB, mask, ctxB);

  // output projection -> f32 d_out
  gemm_wmma_kernel<<<gemmBlocks, blk, 0, stream>>>(ctxB, Wo16, bo, out, NTOK, 2);
  (void)in_sizes; (void)n_in; (void)out_size; (void)ws_size;
}
